// GQAttention_16080357556473
// MI455X (gfx1250) — matrix operand order repaired and checked by reading, not compiled, not run on hardware
//
#include <hip/hip_runtime.h>

// ---------------- problem constants ----------------
static constexpr int Bc  = 2;
static constexpr int Tc  = 2048;
static constexpr int Dc  = 2048;
static constexpr int Hc  = 16;
static constexpr int HKc = 4;
static constexpr int HDc = 128;
static constexpr int NQKV = Hc*HDc + 2*HKc*HDc;   // 3072 (q | k | v)
static constexpr int MROWS = Bc*Tc;               // 4096

#define DEV __device__ __forceinline__
#define NEG_INF (-__builtin_inff())

// ---------------- types ----------------
typedef __bf16          v16bf  __attribute__((ext_vector_type(16)));
typedef float           v8f    __attribute__((ext_vector_type(8)));
typedef unsigned short  u16x4  __attribute__((ext_vector_type(4)));
typedef unsigned short  u16x8  __attribute__((ext_vector_type(8)));
typedef unsigned short  u16x16 __attribute__((ext_vector_type(16)));
typedef unsigned int    u32x4  __attribute__((ext_vector_type(4)));
typedef int             i32x4  __attribute__((ext_vector_type(4)));
typedef int             i32x8  __attribute__((ext_vector_type(8)));

// ---------------- helpers ----------------
DEV unsigned short f2bf(float f) {            // hardware v_cvt_bf16_f32 path
  return __builtin_bit_cast(unsigned short, (__bf16)f);
}
DEV float bf2f(unsigned short b) {
  unsigned int u = ((unsigned int)b) << 16;
  return __builtin_bit_cast(float, u);
}
DEV v8f zero8() {
  v8f z = {0.f,0.f,0.f,0.f,0.f,0.f,0.f,0.f};
  return z;
}
DEV v16bf ld_frag(const unsigned short* p0, const unsigned short* p1) {
  u16x8 a = *(const u16x8*)p0;
  u16x8 b = *(const u16x8*)p1;
  u16x16 r = __builtin_shufflevector(a, b, 0,1,2,3,4,5,6,7,8,9,10,11,12,13,14,15);
  return __builtin_bit_cast(v16bf, r);
}
DEV v8f wmma_bf16(v16bf a, v16bf b, v8f c) {
  return __builtin_amdgcn_wmma_f32_16x16x32_bf16(false, a, false, b, (short)0, c,
                                                 false, false);
}
DEV unsigned lds_off(const void* p) {         // generic addr low 32b = LDS offset
  return (unsigned)(unsigned long long)p;
}

// Tensor Data Mover: 2D bf16 tile (tile_w x tile_h elems, row stride `stride`
// elems in memory) -> LDS at lds_addr, with LDS row padding so the LDS row
// stride becomes tile_w + 8 bf16.  pad codes: interval (DWORDs of data) and
// amount (DWORDs of pad) per ISA D# group1 encoding.
// 6-arg builtin variant (clang-23 / therock-10.0): groups 2,3 are 4-dword,
// plus an extra 8-dword group, all unused for 2D tiles -> zeros.
DEV void tdm_load_2d(const unsigned short* gptr, unsigned lds_addr,
                     unsigned tile_w, unsigned tile_h, unsigned stride,
                     unsigned icode, unsigned acode)
{
  const unsigned long long ga = (unsigned long long)gptr;
  u32x4 g0;
  g0[0] = 1u;                                            // count=1, user mode
  g0[1] = lds_addr;                                      // LDS dest (bytes)
  g0[2] = (unsigned)(ga & 0xffffffffu);                  // global_addr lo
  g0[3] = (unsigned)((ga >> 32) & 0x01ffffffu) | (2u << 30);  // addr hi | type=2
  i32x8 g1;
  g1[0] = (int)((1u << 16)              // data_size = 2 bytes
              | (1u << 20)              // pad_enable
              | (icode << 22)           // pad_interval
              | (acode << 25));         // pad_amount
  g1[1] = (int)(tile_w << 16);          // tensor_dim0[15:0]
  g1[2] = (int)(tile_h << 16);          // tensor_dim0 hi=0 | tensor_dim1[15:0]
  g1[3] = (int)(tile_w << 16);          // tensor_dim1 hi=0 | tile_dim0
  g1[4] = (int)(tile_h & 0xffffu);      // tile_dim1 | tile_dim2=0
  g1[5] = (int)stride;                  // tensor_dim0_stride lo
  g1[6] = 0;
  g1[7] = 0;
  i32x4 z4 = {0,0,0,0};
  i32x8 z8 = {0,0,0,0,0,0,0,0};
  __builtin_amdgcn_tensor_load_to_lds(g0, g1, z4, z4, z8, 0);
}

// =====================================================================
// Prep 1: x (f32) -> bf16, vectorized
// =====================================================================
__global__ __launch_bounds__(256) void cvt_bf16_vec(
    const float* __restrict__ s, unsigned short* __restrict__ d)
{
  const size_t i = (size_t)(blockIdx.x*256 + threadIdx.x) * 4;
  const float4 f = *(const float4*)(s + i);
  u16x4 o; o[0] = f2bf(f.x); o[1] = f2bf(f.y); o[2] = f2bf(f.z); o[3] = f2bf(f.w);
  *(u16x4*)(d + i) = o;
}

// =====================================================================
// Prep 2: weight transpose f32[K=2048][N] -> bf16 dst[n][k] (row len 2048)
// =====================================================================
__global__ __launch_bounds__(256) void wtrans_bf16(
    const float* __restrict__ src, int N, unsigned short* __restrict__ dst)
{
  __shared__ unsigned short tile[32][33];
  const int k0 = blockIdx.y * 32;
  const int n0 = blockIdx.x * 32;
  const int tx = threadIdx.x & 31, ty = threadIdx.x >> 5;   // 32 x 8
#pragma unroll
  for (int i = 0; i < 4; ++i)
    tile[ty + i*8][tx] = f2bf(src[(size_t)(k0 + ty + i*8)*N + n0 + tx]);
  __syncthreads();
#pragma unroll
  for (int i = 0; i < 4; ++i)
    dst[(size_t)(n0 + ty + i*8)*Dc + k0 + tx] = tile[tx][ty + i*8];
}

// =====================================================================
// Kernel 1: fused QKV projection. Y[4096,3072](bf16) = xb @ Wt^T
// TDM double-buffered staging, 8 waves, 128x128 block, K-step 32.
// =====================================================================
__global__ __launch_bounds__(256) void qkv_gemm(
    const unsigned short* __restrict__ xb,   // [4096][2048] bf16
    const unsigned short* __restrict__ Wt,   // [3072][2048] bf16 (n-major)
    unsigned short* __restrict__ Y)
{
  __shared__ unsigned short ldsA[2][128*40];
  __shared__ unsigned short ldsB[2][128*40];

  const int tid  = threadIdx.x;
  const int lane = tid & 31;
  const int wv_  = tid >> 5;
  const int l16  = lane & 15;
  const int hi   = lane >> 4;
  const int bx = blockIdx.x, by = blockIdx.y;
  const int row0 = by * 128, col0 = bx * 128;
  const int wm = wv_ & 3, wn = wv_ >> 2;

  v8f acc[2][4];
#pragma unroll
  for (int mt = 0; mt < 2; ++mt)
#pragma unroll
    for (int nt = 0; nt < 4; ++nt) acc[mt][nt] = zero8();

  const unsigned short* Abase = xb + (size_t)row0*Dc;
  const unsigned short* Bbase = Wt + (size_t)col0*Dc;

  if (wv_ == 0) {                       // prologue: tile 0 into buf 0
    tdm_load_2d(Abase, lds_off(&ldsA[0][0]), 32, 128, Dc, 3, 3);
    tdm_load_2d(Bbase, lds_off(&ldsB[0][0]), 32, 128, Dc, 3, 3);
  }

  const int NK = Dc / 32;               // 64 K-steps
  for (int i = 0; i < NK; ++i) {
    __syncthreads();                    // everyone done with buf[(i+1)&1]
    if (wv_ == 0) {
      if (i + 1 < NK) {
        tdm_load_2d(Abase + (i+1)*32, lds_off(&ldsA[(i+1)&1][0]), 32, 128, Dc, 3, 3);
        tdm_load_2d(Bbase + (i+1)*32, lds_off(&ldsB[(i+1)&1][0]), 32, 128, Dc, 3, 3);
        __builtin_amdgcn_s_wait_tensorcnt(2);   // tile i complete
      } else {
        __builtin_amdgcn_s_wait_tensorcnt(0);
      }
    }
    __syncthreads();                    // buf[i&1] ready
    const unsigned short* A = &ldsA[i&1][0];
    const unsigned short* B = &ldsB[i&1][0];

    v16bf af[2], bf[4];
    const int kb0 = hi ? 8 : 0;
#pragma unroll
    for (int mt = 0; mt < 2; ++mt) {
      const unsigned short* p = &A[(wm*32 + mt*16 + l16)*40 + kb0];
      af[mt] = ld_frag(p, p + 16);
    }
    const int kbb = hi ? 8 : 0;
#pragma unroll
    for (int nt = 0; nt < 4; ++nt) {
      const unsigned short* p = &B[(wn*64 + nt*16 + l16)*40 + kbb];
      bf[nt] = ld_frag(p, p + 16);
    }
#pragma unroll
    for (int mt = 0; mt < 2; ++mt)
#pragma unroll
      for (int nt = 0; nt < 4; ++nt)
        acc[mt][nt] = wmma_bf16(af[mt], bf[nt], acc[mt][nt]);
  }

#pragma unroll
  for (int mt = 0; mt < 2; ++mt)
#pragma unroll
    for (int nt = 0; nt < 4; ++nt)
#pragma unroll
      for (int e = 0; e < 8; ++e) {
        const int r = row0 + wm*32 + mt*16 + e + hi*8;
        const int c = col0 + wn*64 + nt*16 + l16;
        Y[(size_t)r*NQKV + c] = f2bf(acc[mt][nt][e]);
      }
}

// =====================================================================
// Kernel 2: RoPE in place on q and k slices of Y.
// =====================================================================
__global__ __launch_bounds__(256) void rope_kernel(
    unsigned short* __restrict__ Y,
    const float* __restrict__ fcos, const float* __restrict__ fsin)
{
  const int gid = blockIdx.x*256 + threadIdx.x;
  const int row = gid / 1280;
  const int p   = gid - row*1280;
  const int i   = p & 63;
  int col;
  if (p < 1024) col = (p >> 6)*HDc + 2*i;
  else          col = Hc*HDc + ((p - 1024) >> 6)*HDc + 2*i;
  const int t = row & (Tc - 1);
  const float c = fcos[t*64 + i], s = fsin[t*64 + i];
  unsigned short* yp = Y + (size_t)row*NQKV + col;
  const float xr = bf2f(yp[0]), xi = bf2f(yp[1]);
  yp[0] = f2bf(xr*c - xi*s);
  yp[1] = f2bf(xr*s + xi*c);
}

// =====================================================================
// Kernel 3: flash attention.  4 waves = 4 rep-heads of one KV head,
// one 16-query tile.  S^T = K@Q^T, K staged by TDM (double-buffered),
// V transposed manually with packed b32 stores.
// =====================================================================
__global__ __launch_bounds__(128) void attn_fa(
    const unsigned short* __restrict__ Y, unsigned short* __restrict__ AO)
{
  __shared__ unsigned short ldsK[2][32*136];   // [key][d] (+8 pad) via TDM
  __shared__ unsigned short ldsV[128*40];      // [d][key] (+8 pad)
  __shared__ float stats[4*16];

  const int tid  = threadIdx.x;
  const int lane = tid & 31;
  const int w    = tid >> 5;
  const int l16  = lane & 15;
  const int hi   = lane >> 4;
  const int qt = blockIdx.x, kh = blockIdx.y, b = blockIdx.z;
  const int qbase = qt * 16;
  const int h = kh*4 + w;
  const float scale = 0.08838834764831845f;    // 1/sqrt(128)

  // Q fragments (B-matrix of S^T)
  v16bf qf[4];
  {
    const unsigned short* qp =
        Y + (size_t)(b*Tc + qbase + l16)*NQKV + h*HDc + hi*8;
#pragma unroll
    for (int c = 0; c < 4; ++c) qf[c] = ld_frag(qp + c*32, qp + c*32 + 16);
  }

  v8f o[8];
#pragma unroll
  for (int nd = 0; nd < 8; ++nd) o[nd] = zero8();
  float mrow = NEG_INF, lrow = 0.f;

  const unsigned short* Kbase = Y + (size_t)b*Tc*NQKV + Hc*HDc + kh*HDc;
  const int jmax = (qbase + 15) >> 5;

  if (w == 0)
    tdm_load_2d(Kbase, lds_off(&ldsK[0][0]), 128, 32, NQKV, 5, 3);

  for (int j = 0; j <= jmax; ++j) {
    __syncthreads();                          // done with previous tile
    // ---- stage V transposed, packed pairs of keys into b32 stores ----
    {
#pragma unroll
      for (int r = 0; r < 2; ++r) {
        const int u  = tid + 128*r;
        const int kp = u >> 4;                // key pair 0..15
        const int dc = u & 15;                // d chunk (8 elems) 0..15
        const unsigned short* vp =
            Y + (size_t)(b*Tc + j*32 + 2*kp)*NQKV + Hc*HDc + HKc*HDc + kh*HDc + dc*8;
        u16x8 a  = *(const u16x8*)vp;
        u16x8 bq = *(const u16x8*)(vp + NQKV);
#pragma unroll
        for (int i2 = 0; i2 < 8; ++i2) {
          const unsigned wd = (unsigned)a[i2] | ((unsigned)bq[i2] << 16);
          *(unsigned*)&ldsV[(dc*8 + i2)*40 + 2*kp] = wd;
        }
      }
    }
    if (w == 0) {
      if (j + 1 <= jmax) {
        tdm_load_2d(Kbase + (size_t)(j+1)*32*NQKV, lds_off(&ldsK[(j+1)&1][0]),
                    128, 32, NQKV, 5, 3);
        __builtin_amdgcn_s_wait_tensorcnt(1);
      } else {
        __builtin_amdgcn_s_wait_tensorcnt(0);
      }
    }
    __syncthreads();                          // K(buf j) + V ready

    // ---- S^T tiles ----
    const unsigned short* Kt = &ldsK[j&1][0];
    v8f s0 = zero8(), s1 = zero8();
    const int kb0 = hi ? 8 : 0;
#pragma unroll
    for (int c = 0; c < 4; ++c) {
      const unsigned short* p0 = &Kt[(     l16)*136 + c*32 + kb0];
      const unsigned short* p1 = &Kt[(16 + l16)*136 + c*32 + kb0];
      s0 = wmma_bf16(ld_frag(p0, p0 + 16), qf[c], s0);
      s1 = wmma_bf16(ld_frag(p1, p1 + 16), qf[c], s1);
    }

    // ---- scale + causal mask ----
    const int t  = qbase + l16;
    const int sb = j*32 + hi*8;
#pragma unroll
    for (int e = 0; e < 8; ++e) {
      s0[e] = (sb + e      <= t) ? s0[e]*scale : NEG_INF;
      s1[e] = (sb + 16 + e <= t) ? s1[e]*scale : NEG_INF;
    }

    // ---- online softmax ----
    float mt = NEG_INF;
#pragma unroll
    for (int e = 0; e < 8; ++e) mt = fmaxf(mt, fmaxf(s0[e], s1[e]));
    mt = fmaxf(mt, __shfl_xor(mt, 16, 32));
    const float mnew = fmaxf(mrow, mt);
    float rs = 0.f;
#pragma unroll
    for (int e = 0; e < 8; ++e) {
      s0[e] = __expf(s0[e] - mnew);
      s1[e] = __expf(s1[e] - mnew);
      rs += s0[e] + s1[e];
    }
    rs += __shfl_xor(rs, 16, 32);
    const float alpha = __expf(mrow - mnew);
    lrow = lrow*alpha + rs;
    mrow = mnew;

    if (lane < 16) stats[w*16 + lane] = alpha;
    float a8[8];
#pragma unroll
    for (int e = 0; e < 8; ++e) a8[e] = stats[w*16 + hi*8 + e];
#pragma unroll
    for (int nd = 0; nd < 8; ++nd)
#pragma unroll
      for (int e = 0; e < 8; ++e) o[nd][e] *= a8[e];

    // ---- P fragment directly from S^T accumulators ----
    u16x16 pb;
#pragma unroll
    for (int e = 0; e < 8; ++e) { pb[e] = f2bf(s0[e]); pb[e+8] = f2bf(s1[e]); }
    const v16bf pf = __builtin_bit_cast(v16bf, pb);

    // ---- O += P @ V ----
    const int kbv = hi ? 8 : 0;
#pragma unroll
    for (int nd = 0; nd < 8; ++nd) {
      const unsigned short* vpp = &ldsV[(nd*16 + l16)*40 + kbv];
      o[nd] = wmma_bf16(pf, ld_frag(vpp, vpp + 16), o[nd]);
    }
  }

  // ---- finalize ----
  if (lane < 16) stats[w*16 + lane] = lrow;
  float inv[8];
#pragma unroll
  for (int e = 0; e < 8; ++e) inv[e] = 1.f / stats[w*16 + hi*8 + e];
#pragma unroll
  for (int nd = 0; nd < 8; ++nd)
#pragma unroll
    for (int e = 0; e < 8; ++e) {
      const int r = qbase + hi*8 + e;
      AO[(size_t)(b*Tc + r)*(Hc*HDc) + h*HDc + nd*16 + l16] =
          f2bf(o[nd][e] * inv[e]);
    }
}

// =====================================================================
// Kernel 4: out[4096,2048](f32) = AO(bf16) @ Wot^T(bf16), B via TDM.
// =====================================================================
__global__ __launch_bounds__(256) void out_gemm(
    const unsigned short* __restrict__ AO, const unsigned short* __restrict__ Wot,
    float* __restrict__ out)
{
  __shared__ unsigned short ldsB[2][128*40];

  const int tid  = threadIdx.x;
  const int lane = tid & 31;
  const int wv_  = tid >> 5;
  const int l16  = lane & 15;
  const int hi   = lane >> 4;
  const int bx = blockIdx.x, by = blockIdx.y;
  const int row0 = by*128, bcol = bx*128;
  const int wm = wv_ & 3, wn = wv_ >> 2;

  v8f acc[2][4];
#pragma unroll
  for (int mt = 0; mt < 2; ++mt)
#pragma unroll
    for (int nt = 0; nt < 4; ++nt) acc[mt][nt] = zero8();

  const unsigned short* Bbase = Wot + (size_t)bcol*Dc;
  if (wv_ == 0)
    tdm_load_2d(Bbase, lds_off(&ldsB[0][0]), 32, 128, Dc, 3, 3);

  const int NK = (Hc*HDc) / 32;
  for (int i = 0; i < NK; ++i) {
    __syncthreads();
    if (wv_ == 0) {
      if (i + 1 < NK) {
        tdm_load_2d(Bbase + (i+1)*32, lds_off(&ldsB[(i+1)&1][0]), 32, 128, Dc, 3, 3);
        __builtin_amdgcn_s_wait_tensorcnt(1);
      } else {
        __builtin_amdgcn_s_wait_tensorcnt(0);
      }
    }
    __syncthreads();
    const unsigned short* B = &ldsB[i&1][0];
    const int k0 = i*32;

    v16bf af[2], bf[4];
    const int kb0 = hi ? 8 : 0;
#pragma unroll
    for (int mt = 0; mt < 2; ++mt) {
      const unsigned short* ap =
          AO + (size_t)(row0 + wm*32 + mt*16 + l16)*(Hc*HDc) + k0 + kb0;
      af[mt] = ld_frag(ap, ap + 16);
    }
    const int kbb = hi ? 8 : 0;
#pragma unroll
    for (int nt = 0; nt < 4; ++nt) {
      const unsigned short* p = &B[(wn*64 + nt*16 + l16)*40 + kbb];
      bf[nt] = ld_frag(p, p + 16);
    }
#pragma unroll
    for (int mt = 0; mt < 2; ++mt)
#pragma unroll
      for (int nt = 0; nt < 4; ++nt)
        acc[mt][nt] = wmma_bf16(af[mt], bf[nt], acc[mt][nt]);
  }

#pragma unroll
  for (int mt = 0; mt < 2; ++mt)
#pragma unroll
    for (int nt = 0; nt < 4; ++nt)
#pragma unroll
      for (int e = 0; e < 8; ++e) {
        const int r = row0 + wm*32 + mt*16 + e + hi*8;
        const int c = bcol + wn*64 + nt*16 + l16;
        out[(size_t)r*Dc + c] = acc[mt][nt][e];
      }
}

// =====================================================================
extern "C" void kernel_launch(void* const* d_in, const int* in_sizes, int n_in,
                              void* d_out, int out_size, void* d_ws, size_t ws_size,
                              hipStream_t stream) {
  const float* x    = (const float*)d_in[0];
  const float* fcos = (const float*)d_in[1];
  const float* fsin = (const float*)d_in[2];
  // d_in[3] = mask (causality applied analytically)
  const float* wq   = (const float*)d_in[4];
  const float* wk   = (const float*)d_in[5];
  const float* wv   = (const float*)d_in[6];
  const float* wo   = (const float*)d_in[7];
  float* out = (float*)d_out;

  unsigned short* Y   = (unsigned short*)d_ws;              // [4096][3072]
  unsigned short* AO  = Y   + (size_t)MROWS*NQKV;           // [4096][2048]
  unsigned short* xb  = AO  + (size_t)MROWS*Dc;             // [4096][2048]
  unsigned short* Wt  = xb  + (size_t)MROWS*Dc;             // [3072][2048]
  unsigned short* Wot = Wt  + (size_t)NQKV*Dc;              // [2048][2048]

  // prep: bf16 conversion + weight transposes (one-time, bandwidth-trivial)
  cvt_bf16_vec<<<(MROWS*Dc)/1024, 256, 0, stream>>>(x, xb);
  wtrans_bf16<<<dim3(64, 64), 256, 0, stream>>>(wq, 2048, Wt);
  wtrans_bf16<<<dim3(16, 64), 256, 0, stream>>>(wk,  512, Wt + (size_t)2048*Dc);
  wtrans_bf16<<<dim3(16, 64), 256, 0, stream>>>(wv,  512, Wt + (size_t)2560*Dc);
  wtrans_bf16<<<dim3(64, 64), 256, 0, stream>>>(wo, 2048, Wot);

  qkv_gemm<<<dim3(NQKV/128, MROWS/128), 256, 0, stream>>>(xb, Wt, Y);
  rope_kernel<<<(MROWS*1280)/256, 256, 0, stream>>>(Y, fcos, fsin);
  attn_fa<<<dim3(Tc/16, HKc, Bc), 128, 0, stream>>>(Y, AO);
  out_gemm<<<dim3(Dc/128, MROWS/128), 256, 0, stream>>>(AO, Wot, out);
}
